// PairwiseScore_56289841381909
// MI455X (gfx1250) — compile-verified, matrix operand order checked by reading, not run on hardware
//
#include <hip/hip_runtime.h>
#include <hip/hip_bf16.h>
#include <math.h>

typedef _Float16 half16 __attribute__((ext_vector_type(16)));
typedef _Float16 half8  __attribute__((ext_vector_type(8)));
typedef _Float16 half2v __attribute__((ext_vector_type(2)));
typedef float    float8 __attribute__((ext_vector_type(8)));

#define GDIM  512
#define HID   150
#define GIJ   1596   // 3*512 + 60
#define KP1   1600   // GIJ padded to multiple of 32
#define NP    160    // HID padded to multiple of 16
#define NT    10     // N tiles of 16 (160/16)
#define KT1   50     // K tiles of 32 for layer 1 (1600/32)
#define KT2   5      // K tiles of 32 for layer 2 (160/32)
#define KMAXC 31     // KMAX+1 output columns
#define EMB   20
#define MT    2      // M tiles (16 rows each) per wave -> B-fragment reuse x2
#define ROWS  (MT * 16)
#define WAVES 2      // waves per block, N dimension split across waves
#define NTW   (NT / WAVES)   // N tiles per wave (5)
#define KC    800    // staged K columns per chunk (2 chunks cover KP1)
#define KTC   (KC / 32)      // kt iterations per chunk (25)

#define FRAG1 (KT1 * 32 * 16)   // halfs per N-tile of packed W1 (25600)
#define FRAG2 (KT2 * 32 * 16)   // halfs per N-tile of packed W2 (2560)

// ---------------------------------------------------------------------------
// Map WMMA 16-bit A/B fragment element (lane, e) -> K index within a 32-K tile
// (per CDNA5 ISA 16-bit A-matrix 16x32 layout; B uses same striping with N).
__device__ __forceinline__ int frag_k(int lane, int e) {
    int v = e >> 1, h = e & 1, khalf = lane >> 4;
    return (v < 4) ? (khalf * 8 + v * 2 + h)
                   : (16 + khalf * 8 + (v - 4) * 2 + h);
}

// Feature vector column c of a pair row: [i_g | j_g | i_g*j_g | phi | 0-pad]
__device__ __forceinline__ float feature_col(
    int c, const float* __restrict__ gi, const float* __restrict__ gj,
    const float* __restrict__ de, const float* __restrict__ ge,
    const float* __restrict__ se) {
    if (c < GDIM)     return gi[c];
    if (c < 2 * GDIM) return gj[c - GDIM];
    if (c < 3 * GDIM) { int q = c - 2 * GDIM; return gi[q] * gj[q]; }
    int q = c - 3 * GDIM;
    if (q < EMB)      return de[q];
    if (q < 2 * EMB)  return ge[q - EMB];
    if (q < 3 * EMB)  return se[q - 2 * EMB];
    return 0.f;
}

// ---------------------------------------------------------------------------
// Pack W1 [1596x150] and W2 [150x150] (row-major f32) into fragment-major f16:
// dst[((nt*KT + kt)*32 + lane)*16 + e]  with zero padding outside K/N.
__global__ __launch_bounds__(256) void pack_weights_kernel(
    const float* __restrict__ W1, const float* __restrict__ W2,
    _Float16* __restrict__ W1p, _Float16* __restrict__ W2p) {
    const int T1 = NT * FRAG1;
    const int T2 = NT * FRAG2;
    int idx = blockIdx.x * blockDim.x + threadIdx.x;
    if (idx < T1) {
        int e = idx & 15, lane = (idx >> 4) & 31;
        int frag = idx >> 9;            // nt*KT1 + kt
        int kt = frag % KT1, nt = frag / KT1;
        int n = nt * 16 + (lane & 15);
        int k = kt * 32 + frag_k(lane, e);
        float v = (k < GIJ && n < HID) ? W1[k * HID + n] : 0.f;
        W1p[idx] = (_Float16)v;
    } else if (idx < T1 + T2) {
        int j = idx - T1;
        int e = j & 15, lane = (j >> 4) & 31;
        int frag = j >> 9;              // nt*KT2 + kt
        int kt = frag % KT2, nt = frag / KT2;
        int n = nt * 16 + (lane & 15);
        int k = kt * 32 + frag_k(lane, e);
        float v = (k < HID && n < HID) ? W2[k * HID + n] : 0.f;
        W2p[j] = (_Float16)v;
    }
}

// ---------------------------------------------------------------------------
// One 2-wave block per 32-row tile of P. Each wave owns 5 N-tiles (wave index
// scalarized via readfirstlane so B addressing stays SGPR-base + immediate);
// both waves share the K-chunked feature stage in LDS. Each B fragment feeds
// 2 WMMAs (M-tile reuse). Layer-1 K loop runs in 2 passes over 800-column
// chunks with persistent f32 accumulators: LDS = 61.4 KB -> ~10 waves/WGP.
__global__ __launch_bounds__(WAVES * 32) void pair_mlp_kernel(
    const float* __restrict__ g_i, const float* __restrict__ mention_scores,
    const float* __restrict__ dist_emb, const float* __restrict__ genre_emb,
    const float* __restrict__ speaker_emb,
    const float* __restrict__ b1, const float* __restrict__ b2,
    const float* __restrict__ W3, const float* __restrict__ b3,
    const int* __restrict__ mention_ids, const int* __restrict__ antecedent_ids,
    const int* __restrict__ dist_idx, const int* __restrict__ genre_idx,
    const int* __restrict__ speaker_idx,
    const int* __restrict__ seg_ids, const int* __restrict__ pos_in_seg,
    const _Float16* __restrict__ W1p, const _Float16* __restrict__ W2p,
    float* __restrict__ dense, int P) {

    extern __shared__ __align__(32) char smem[];
    _Float16 (*sP)[KC] = (_Float16 (*)[KC])smem;                     // [32][800]
    _Float16 (*sH)[NP] = (_Float16 (*)[NP])(smem + ROWS * KC * 2);   // [32][160]

    const int tile = blockIdx.x;
    const int tid  = threadIdx.x;
    const int lane = tid & 31;
    // tid>>5 is wave-invariant; readfirstlane makes it provably scalar so the
    // per-wave weight base lands in SGPRs (saddr+imm addressing in the loop).
    const int wave = __builtin_amdgcn_readfirstlane(tid >> 5);
    const int arow  = lane & 15;
    const int khalf = lane >> 4;

    const _Float16* __restrict__ W1w = W1p + (size_t)wave * NTW * FRAG1;
    const _Float16* __restrict__ W2w = W2p + (size_t)wave * NTW * FRAG2;

    float8 acc[MT][NTW];
#pragma unroll
    for (int mt = 0; mt < MT; ++mt)
#pragma unroll
        for (int nl = 0; nl < NTW; ++nl)
#pragma unroll
            for (int i = 0; i < 8; ++i) acc[mt][nl][i] = 0.f;

    // ---- layer 1: two K-chunks of 800 columns, persistent accumulators -----
    for (int kc = 0; kc < 2; ++kc) {
        // fill chunk [kc*KC, kc*KC+KC) of the feature tile
        for (int r = 0; r < ROWS; ++r) {
            int p = tile * ROWS + r;
            _Float16* sPr = sP[r];
            if (p < P) {
                const float* gi = g_i + (size_t)mention_ids[p] * GDIM;
                const float* gj = g_i + (size_t)antecedent_ids[p] * GDIM;
                const float* de = dist_emb    + dist_idx[p]    * EMB;
                const float* ge = genre_emb   + genre_idx[p]   * EMB;
                const float* se = speaker_emb + speaker_idx[p] * EMB;
                for (int cc = tid * 2; cc < KC; cc += WAVES * 64) {
                    int c = kc * KC + cc;
                    float v0 = feature_col(c,     gi, gj, de, ge, se);
                    float v1 = feature_col(c + 1, gi, gj, de, ge, se);
                    *(half2v*)&sPr[cc] = (half2v){(_Float16)v0, (_Float16)v1};
                }
            } else {
                half8 z = (half8)(_Float16)0.f;
                for (int cc = tid * 8; cc < KC; cc += WAVES * 256)
                    *(half8*)&sPr[cc] = z;
            }
        }
        __syncthreads();

        for (int ktl = 0; ktl < KTC; ++ktl) {
            int kt = kc * KTC + ktl;
            half16 A[MT];
#pragma unroll
            for (int mt = 0; mt < MT; ++mt) {
                const half8* a1 = (const half8*)&sP[mt * 16 + arow][ktl * 32 + khalf * 8];
                const half8* a2 = (const half8*)&sP[mt * 16 + arow][ktl * 32 + 16 + khalf * 8];
#pragma unroll
                for (int e = 0; e < 8; ++e) { A[mt][e] = (*a1)[e]; A[mt][8 + e] = (*a2)[e]; }
            }
            // scalar base advances with kt; per-lane part is lane*32B;
            // per-nl part is a compile-time immediate (nl * 51200B).
            const _Float16* bk = W1w + (((size_t)kt * 32 + lane) << 4);
#pragma unroll
            for (int nl = 0; nl < NTW; ++nl) {
                const half16 B = *(const half16*)(bk + (size_t)nl * FRAG1);
#pragma unroll
                for (int mt = 0; mt < MT; ++mt)
                    acc[mt][nl] = __builtin_amdgcn_wmma_f32_16x16x32_f16(
                        false, A[mt], false, B, (short)0, acc[mt][nl], false, false);
            }
        }
        __syncthreads();   // chunk consumed before refill
    }

    // bias + relu -> sH (f16), zero in the N-pad region
#pragma unroll
    for (int mt = 0; mt < MT; ++mt)
#pragma unroll
        for (int nl = 0; nl < NTW; ++nl) {
            int n  = (wave * NTW + nl) * 16 + (lane & 15);
            int mb = mt * 16 + (lane >> 4) * 8;
            float bias = (n < HID) ? b1[n] : 0.f;
#pragma unroll
            for (int r8 = 0; r8 < 8; ++r8) {
                float v = acc[mt][nl][r8] + bias;
                v = v > 0.f ? v : 0.f;
                if (n >= HID) v = 0.f;
                sH[mb + r8][n] = (_Float16)v;
            }
        }
    __syncthreads();

    // ---- layer 2: [32x160] x [160x160] via WMMA ----------------------------
    float8 acc2[MT][NTW];
#pragma unroll
    for (int mt = 0; mt < MT; ++mt)
#pragma unroll
        for (int nl = 0; nl < NTW; ++nl)
#pragma unroll
            for (int i = 0; i < 8; ++i) acc2[mt][nl][i] = 0.f;

#pragma unroll
    for (int kt = 0; kt < KT2; ++kt) {
        half16 A[MT];
#pragma unroll
        for (int mt = 0; mt < MT; ++mt) {
            const half8* a1 = (const half8*)&sH[mt * 16 + arow][kt * 32 + khalf * 8];
            const half8* a2 = (const half8*)&sH[mt * 16 + arow][kt * 32 + 16 + khalf * 8];
#pragma unroll
            for (int e = 0; e < 8; ++e) { A[mt][e] = (*a1)[e]; A[mt][8 + e] = (*a2)[e]; }
        }
        const _Float16* bk = W2w + (((size_t)kt * 32 + lane) << 4);
#pragma unroll
        for (int nl = 0; nl < NTW; ++nl) {
            const half16 B = *(const half16*)(bk + (size_t)nl * FRAG2);
#pragma unroll
            for (int mt = 0; mt < MT; ++mt)
                acc2[mt][nl] = __builtin_amdgcn_wmma_f32_16x16x32_f16(
                    false, A[mt], false, B, (short)0, acc2[mt][nl], false, false);
        }
    }
    __syncthreads();  // all layer-2 reads of sH done before overwrite

#pragma unroll
    for (int mt = 0; mt < MT; ++mt)
#pragma unroll
        for (int nl = 0; nl < NTW; ++nl) {
            int n  = (wave * NTW + nl) * 16 + (lane & 15);
            int mb = mt * 16 + (lane >> 4) * 8;
            float bias = (n < HID) ? b2[n] : 0.f;
#pragma unroll
            for (int r8 = 0; r8 < 8; ++r8) {
                float v = acc2[mt][nl][r8] + bias;
                v = v > 0.f ? v : 0.f;
                if (n >= HID) v = 0.f;
                sH[mb + r8][n] = (_Float16)v;
            }
        }
    __syncthreads();

    // ---- layer 3 + coref score + scatter into dense ------------------------
    if (tid < ROWS) {
        int p = tile * ROWS + tid;   // one row per lane of wave 0
        if (p < P) {
            float s = b3[0];
            for (int k = 0; k < HID; ++k) s += (float)sH[tid][k] * W3[k];
            float c = mention_scores[mention_ids[p]] +
                      mention_scores[antecedent_ids[p]] + s;
            dense[(size_t)seg_ids[p] * KMAXC + pos_in_seg[p]] = c;
        }
    }
}

// ---------------------------------------------------------------------------
// Per-segment softmax over [len coref scores, epsilon=0], 1000-pad beyond len.
__global__ __launch_bounds__(32) void seg_softmax_kernel(
    const float* __restrict__ dense, const int* __restrict__ seg_lengths,
    float* __restrict__ out, int S) {
    int s = blockIdx.x;
    if (s >= S) return;
    int t = threadIdx.x;
    int len = seg_lengths[s];

    float val;
    if (t < len)        val = dense[(size_t)s * KMAXC + t];
    else if (t == len)  val = 0.f;          // epsilon slot
    else                val = -INFINITY;    // unused slot

    float m = val;
#pragma unroll
    for (int off = 16; off > 0; off >>= 1) m = fmaxf(m, __shfl_xor(m, off, 32));

    float e = (t <= len) ? __expf(val - m) : 0.f;
    float sum = e;
#pragma unroll
    for (int off = 16; off > 0; off >>= 1) sum += __shfl_xor(sum, off, 32);

    if (t < KMAXC)
        out[(size_t)s * KMAXC + t] = (t > len) ? 1000.f : (e / sum);
}

// ---------------------------------------------------------------------------
extern "C" void kernel_launch(void* const* d_in, const int* in_sizes, int n_in,
                              void* d_out, int out_size, void* d_ws, size_t ws_size,
                              hipStream_t stream) {
    const float* g_i            = (const float*)d_in[0];
    const float* mention_scores = (const float*)d_in[1];
    const float* dist_emb       = (const float*)d_in[2];
    const float* genre_emb      = (const float*)d_in[3];
    const float* speaker_emb    = (const float*)d_in[4];
    const float* W1             = (const float*)d_in[5];
    const float* b1             = (const float*)d_in[6];
    const float* W2             = (const float*)d_in[7];
    const float* b2             = (const float*)d_in[8];
    const float* W3             = (const float*)d_in[9];
    const float* b3             = (const float*)d_in[10];
    const int* mention_ids      = (const int*)d_in[11];
    const int* antecedent_ids   = (const int*)d_in[12];
    const int* dist_idx         = (const int*)d_in[13];
    const int* genre_idx        = (const int*)d_in[14];
    const int* speaker_idx      = (const int*)d_in[15];
    const int* seg_ids          = (const int*)d_in[16];
    const int* pos_in_seg       = (const int*)d_in[17];
    const int* seg_lengths      = (const int*)d_in[18];

    const int P = in_sizes[11];
    const int S = in_sizes[18];

    const int T1 = NT * FRAG1;   // 256000 f16
    const int T2 = NT * FRAG2;   //  25600 f16
    _Float16* W1p = (_Float16*)d_ws;
    _Float16* W2p = W1p + T1;
    float* dense  = (float*)((char*)d_ws + (size_t)(T1 + T2) * sizeof(_Float16));

    int packThreads = T1 + T2;
    pack_weights_kernel<<<(packThreads + 255) / 256, 256, 0, stream>>>(W1, W2, W1p, W2p);

    int tiles = (P + ROWS - 1) / ROWS;
    size_t smemBytes = (size_t)ROWS * KC * sizeof(_Float16) +
                       (size_t)ROWS * NP * sizeof(_Float16);   // 61,440 B
    pair_mlp_kernel<<<tiles, WAVES * 32, smemBytes, stream>>>(
        g_i, mention_scores, dist_emb, genre_emb, speaker_emb,
        b1, b2, W3, b3,
        mention_ids, antecedent_ids, dist_idx, genre_idx, speaker_idx,
        seg_ids, pos_in_seg, W1p, W2p, dense, P);

    seg_softmax_kernel<<<S, 32, 0, stream>>>(dense, seg_lengths, (float*)d_out, S);
}
